// Attention_11759620456792
// MI455X (gfx1250) — matrix operand order repaired and checked by reading, not compiled, not run on hardware
//
#include <hip/hip_runtime.h>
#include <hip/hip_bf16.h>

typedef __bf16 bf16;
typedef __attribute__((ext_vector_type(16))) __bf16 v16bf;
typedef __attribute__((ext_vector_type(8)))  __bf16 v8bf;
typedef __attribute__((ext_vector_type(8)))  float  v8f;
typedef __attribute__((ext_vector_type(4)))  int    v4i;

union BF16x16 { v16bf v; v8bf h[2]; };

#ifndef __has_builtin
#define __has_builtin(x) 0
#endif

#if __has_builtin(__builtin_amdgcn_global_load_async_to_lds_b128)
#define HAVE_ASYNC_LDS 1
typedef __attribute__((address_space(1))) v4i as1_v4i;
typedef __attribute__((address_space(3))) v4i as3_v4i;
#else
#define HAVE_ASYNC_LDS 0
#endif

// 16-byte global -> LDS copy. Async DMA (ASYNCcnt) on CDNA5 when available.
__device__ __forceinline__ void copy16_g2l(const bf16* g, bf16* l) {
#if HAVE_ASYNC_LDS
  __builtin_amdgcn_global_load_async_to_lds_b128((as1_v4i*)g, (as3_v4i*)l, 0, 0);
#else
  *(v8bf*)l = *(const v8bf*)g;
#endif
}

__device__ __forceinline__ void async_wait_all() {
#if HAVE_ASYNC_LDS
#if __has_builtin(__builtin_amdgcn_s_wait_asynccnt)
  __builtin_amdgcn_s_wait_asynccnt(0);
#else
  asm volatile("s_wait_asynccnt 0" ::: "memory");
#endif
#endif
}

// Stage a 128-row x 32-col bf16 tile (g points at (row0,k0), leading dim ldg).
__device__ __forceinline__ void stage_128x32(const bf16* __restrict__ g, int ldg,
                                             bf16* l, int tid) {
#pragma unroll
  for (int u = tid; u < 512; u += 256) {
    int r = u >> 2, c = (u & 3) * 8;
    copy16_g2l(g + (size_t)r * ldg + c, l + r * 32 + c);
  }
}

// Stage a 64x64 bf16 tile (leading dim ldg), into LDS with row stride 64.
__device__ __forceinline__ void stage_64x64(const bf16* __restrict__ g, int ldg,
                                            bf16* l, int tid) {
#pragma unroll
  for (int u = tid; u < 512; u += 256) {
    int r = u >> 3, c = (u & 7) * 8;
    copy16_g2l(g + (size_t)r * ldg + c, l + r * 64 + c);
  }
}

// A-fragment (16x32 bf16, MxK) for v_wmma_f32_16x16x32_bf16.
__device__ __forceinline__ v16bf load_a_frag(const bf16* base, int ld, int row,
                                             int k0, int lane) {
  const int h = lane >> 4;
  const int m = lane & 15;
  const bf16* p = base + (size_t)(row + m) * ld + k0;
  BF16x16 f;
  f.h[0] = *(const v8bf*)(p + h * 8);
  f.h[1] = *(const v8bf*)(p + 16 + h * 8);
  return f.v;
}

// B-fragment (32x16 bf16, KxN), B stored N-major with K contiguous ([n][k]).
__device__ __forceinline__ v16bf load_b_frag(const bf16* base, int ld, int col,
                                             int k0, int lane) {
  const int h = lane >> 4;
  const int n = lane & 15;
  const bf16* p = base + (size_t)(col + n) * ld + k0;
  BF16x16 f;
  f.h[0] = *(const v8bf*)(p + h * 8);
  f.h[1] = *(const v8bf*)(p + 16 + h * 8);
  return f.v;
}

__device__ __forceinline__ v8f wmma_bf16(v16bf a, v16bf b, v8f c) {
  return __builtin_amdgcn_wmma_f32_16x16x32_bf16(false, a, false, b, (short)0, c,
                                                 false, false);
}

// ---------------------------------------------------------------- converts
__global__ __launch_bounds__(256) void cvt_kernel(const float* __restrict__ in,
                                                  bf16* __restrict__ out, int n) {
  int i = blockIdx.x * 256 + threadIdx.x;
  if (i < n) out[i] = (bf16)in[i];
}

// in: [K][N] f32 row-major  ->  out: [N][K] bf16 (K contiguous)
__global__ __launch_bounds__(256) void tcvt_kernel(const float* __restrict__ in,
                                                   bf16* __restrict__ out,
                                                   int K, int N) {
  int i = blockIdx.x * 256 + threadIdx.x;
  if (i < K * N) {
    int k = i / N, n = i - k * N;
    out[(size_t)n * K + k] = (bf16)in[i];
  }
}

// ------------------------------------------------------- QKV projection GEMM
// A: Xb [8192][1024] bf16,  Bt: WqkvT [3072][1024] bf16.
// Epilogue scatters into Q [bh][n][64] (scaled by 0.125*log2e), K [bh][n][64],
// Vt [bh][64][n]. Double-buffered LDS + async copies.
__global__ __launch_bounds__(256) void qkv_gemm_kernel(
    const bf16* __restrict__ A, const bf16* __restrict__ Bt,
    bf16* __restrict__ Qd, bf16* __restrict__ Kd, bf16* __restrict__ Vt) {
  __shared__ __align__(16) bf16 As[2][128 * 32];
  __shared__ __align__(16) bf16 Bs[2][128 * 32];
  const int tid  = threadIdx.x;
  const int lane = tid & 31;
  const int wave = tid >> 5;
  const int wm = wave & 3;   // 4 waves along M (32 rows each)
  const int wn = wave >> 2;  // 2 waves along N (64 cols each)
  const int bm = blockIdx.y * 128;
  const int bn = blockIdx.x * 128;
  const int K = 1024;
  const bf16* Abase = A + (size_t)bm * K;
  const bf16* Bbase = Bt + (size_t)bn * K;

  stage_128x32(Abase, K, As[0], tid);
  stage_128x32(Bbase, K, Bs[0], tid);
  async_wait_all();
  __syncthreads();

  v8f acc[2][4] = {};
  for (int k0 = 0; k0 < K; k0 += 32) {
    const int cur = (k0 >> 5) & 1;
    if (k0 + 32 < K) {
      stage_128x32(Abase + k0 + 32, K, As[cur ^ 1], tid);
      stage_128x32(Bbase + k0 + 32, K, Bs[cur ^ 1], tid);
    }
    v16bf af[2], bfr[4];
#pragma unroll
    for (int i = 0; i < 2; i++) af[i] = load_a_frag(As[cur], 32, wm * 32 + i * 16, 0, lane);
#pragma unroll
    for (int j = 0; j < 4; j++) bfr[j] = load_b_frag(Bs[cur], 32, wn * 64 + j * 16, 0, lane);
#pragma unroll
    for (int i = 0; i < 2; i++)
#pragma unroll
      for (int j = 0; j < 4; j++)
        acc[i][j] = wmma_bf16(af[i], bfr[j], acc[i][j]);
    async_wait_all();
    __syncthreads();
  }

  const int h = lane >> 4, ln = lane & 15;
  const float SCALE_Q = 0.125f * 1.44269504088896340736f;  // fold log2e for exp2
#pragma unroll
  for (int i = 0; i < 2; i++)
#pragma unroll
    for (int j = 0; j < 4; j++)
#pragma unroll
      for (int e = 0; e < 8; e++) {
        int row = bm + wm * 32 + i * 16 + e + h * 8;  // 0..8191  (= b*2048+n)
        int col = bn + wn * 64 + j * 16 + ln;         // 0..3071
        float v = acc[i][j][e];
        int which = col >> 10;        // 0=q 1=k 2=v
        int inner = col & 1023;
        int head = inner >> 6;
        int dcol = inner & 63;
        int batch = row >> 11;
        int nq = row & 2047;
        size_t bh = (size_t)batch * 16 + head;
        if (which == 0)      Qd[(bh * 2048 + nq) * 64 + dcol] = (bf16)(v * SCALE_Q);
        else if (which == 1) Kd[(bh * 2048 + nq) * 64 + dcol] = (bf16)v;
        else                 Vt[(bh * 64 + dcol) * 2048 + nq] = (bf16)v;
      }
}

// ----------------------------------------------------------- flash attention
// grid: (16 row-blocks, 64 bh). 8 waves x 16 query rows. Keys in chunks of 64,
// double-buffered in LDS with async copies; softmax in exp2 domain.
__global__ __launch_bounds__(256) void attn_kernel(const bf16* __restrict__ Qd,
                                                   const bf16* __restrict__ Kd,
                                                   const bf16* __restrict__ Vt,
                                                   bf16* __restrict__ Ob) {
  __shared__ __align__(16) bf16 Ks[2][64 * 64];     // [key][d]
  __shared__ __align__(16) bf16 Vs[2][64 * 64];     // [d][key]
  __shared__ __align__(16) bf16 Ps[8][16 * 64];     // per-wave P [16q][64key]
  const int tid  = threadIdx.x;
  const int lane = tid & 31;
  const int wave = tid >> 5;
  const int bh = blockIdx.y;
  const int q0 = blockIdx.x * 128 + wave * 16;
  const bf16* Qb = Qd + (size_t)bh * 2048 * 64;
  const bf16* Kb = Kd + (size_t)bh * 2048 * 64;
  const bf16* Vb = Vt + (size_t)bh * 64 * 2048;
  const int h = lane >> 4, ln = lane & 15;

  // Q fragments (d=64 -> two K-steps of 32), loaded once from global
  v16bf qf[2];
  qf[0] = load_a_frag(Qb, 64, q0, 0, lane);
  qf[1] = load_a_frag(Qb, 64, q0, 32, lane);

  v8f o[4] = {};          // 16q x 64d accumulator (4 N-tiles)
  float mrow[8], lrow[8]; // per-row online-softmax stats (row = e + 8*h)
#pragma unroll
  for (int e = 0; e < 8; e++) { mrow[e] = -1e30f; lrow[e] = 0.f; }
  bf16* myP = &Ps[wave][0];

  stage_64x64(Kb, 64, Ks[0], tid);
  stage_64x64(Vb, 2048, Vs[0], tid);
  async_wait_all();
  __syncthreads();

  for (int kc = 0; kc < 2048; kc += 64) {
    const int cur = (kc >> 6) & 1;
    if (kc + 64 < 2048) {
      stage_64x64(Kb + (size_t)(kc + 64) * 64, 64, Ks[cur ^ 1], tid);
      stage_64x64(Vb + (kc + 64), 2048, Vs[cur ^ 1], tid);
    }

    // S = Q K^T : 4 tiles of 16 keys, 2 WMMAs (k-steps) each
    v8f s[4];
#pragma unroll
    for (int t = 0; t < 4; t++) {
      v8f z = {};
      v16bf b0 = load_b_frag(Ks[cur], 64, t * 16, 0, lane);
      v16bf b1 = load_b_frag(Ks[cur], 64, t * 16, 32, lane);
      z = wmma_bf16(qf[0], b0, z);
      z = wmma_bf16(qf[1], b1, z);
      s[t] = z;
    }

    // online softmax, exp2 domain (row = e + 8*h; row spans the 16-lane half)
#pragma unroll
    for (int e = 0; e < 8; e++) {
      float cm = fmaxf(fmaxf(s[0][e], s[1][e]), fmaxf(s[2][e], s[3][e]));
      cm = fmaxf(cm, __shfl_xor(cm, 1, 32));
      cm = fmaxf(cm, __shfl_xor(cm, 2, 32));
      cm = fmaxf(cm, __shfl_xor(cm, 4, 32));
      cm = fmaxf(cm, __shfl_xor(cm, 8, 32));
      float nm = fmaxf(mrow[e], cm);
      float alpha = exp2f(mrow[e] - nm);
      mrow[e] = nm;
      lrow[e] *= alpha;
#pragma unroll
      for (int t = 0; t < 4; t++) o[t][e] = o[t][e] * alpha;
      float rs = 0.f;
#pragma unroll
      for (int t = 0; t < 4; t++) {
        float p = exp2f(s[t][e] - nm);
        s[t][e] = p;
        rs += p;
      }
      rs += __shfl_xor(rs, 1, 32);
      rs += __shfl_xor(rs, 2, 32);
      rs += __shfl_xor(rs, 4, 32);
      rs += __shfl_xor(rs, 8, 32);
      lrow[e] += rs;
    }

    // C/D layout -> A layout via per-wave LDS staging
#pragma unroll
    for (int t = 0; t < 4; t++)
#pragma unroll
      for (int e = 0; e < 8; e++)
        myP[(e + h * 8) * 64 + t * 16 + ln] = (bf16)s[t][e];

    // O += P V : 2 k-steps x 4 d-tiles
#pragma unroll
    for (int ks = 0; ks < 2; ks++) {
      v16bf pa = load_a_frag(myP, 64, 0, ks * 32, lane);
#pragma unroll
      for (int t = 0; t < 4; t++) {
        v16bf vb = load_b_frag(Vs[cur], 64, t * 16, ks * 32, lane);
        o[t] = wmma_bf16(pa, vb, o[t]);
      }
    }
    async_wait_all();
    __syncthreads();
  }

  const int batch = bh >> 4, head = bh & 15;
#pragma unroll
  for (int t = 0; t < 4; t++)
#pragma unroll
    for (int e = 0; e < 8; e++) {
      int row = q0 + e + h * 8;
      int dcol = t * 16 + ln;
      float val = o[t][e] / lrow[e];
      Ob[((size_t)batch * 2048 + row) * 1024 + head * 64 + dcol] = (bf16)val;
    }
}

// --------------------------------------------------------- output projection
// A: Ob [8192][1024] bf16, Bt: WoutT [1024][1024] bf16, out f32 + bias.
__global__ __launch_bounds__(256) void out_gemm_kernel(
    const bf16* __restrict__ A, const bf16* __restrict__ Bt,
    const float* __restrict__ bias, float* __restrict__ out) {
  __shared__ __align__(16) bf16 As[2][128 * 32];
  __shared__ __align__(16) bf16 Bs[2][128 * 32];
  const int tid  = threadIdx.x;
  const int lane = tid & 31;
  const int wave = tid >> 5;
  const int wm = wave & 3;
  const int wn = wave >> 2;
  const int bm = blockIdx.y * 128;
  const int bn = blockIdx.x * 128;
  const int K = 1024;
  const bf16* Abase = A + (size_t)bm * K;
  const bf16* Bbase = Bt + (size_t)bn * K;

  stage_128x32(Abase, K, As[0], tid);
  stage_128x32(Bbase, K, Bs[0], tid);
  async_wait_all();
  __syncthreads();

  v8f acc[2][4] = {};
  for (int k0 = 0; k0 < K; k0 += 32) {
    const int cur = (k0 >> 5) & 1;
    if (k0 + 32 < K) {
      stage_128x32(Abase + k0 + 32, K, As[cur ^ 1], tid);
      stage_128x32(Bbase + k0 + 32, K, Bs[cur ^ 1], tid);
    }
    v16bf af[2], bfr[4];
#pragma unroll
    for (int i = 0; i < 2; i++) af[i] = load_a_frag(As[cur], 32, wm * 32 + i * 16, 0, lane);
#pragma unroll
    for (int j = 0; j < 4; j++) bfr[j] = load_b_frag(Bs[cur], 32, wn * 64 + j * 16, 0, lane);
#pragma unroll
    for (int i = 0; i < 2; i++)
#pragma unroll
      for (int j = 0; j < 4; j++)
        acc[i][j] = wmma_bf16(af[i], bfr[j], acc[i][j]);
    async_wait_all();
    __syncthreads();
  }

  const int h = lane >> 4, ln = lane & 15;
#pragma unroll
  for (int i = 0; i < 2; i++)
#pragma unroll
    for (int j = 0; j < 4; j++)
#pragma unroll
      for (int e = 0; e < 8; e++) {
        int row = bm + wm * 32 + i * 16 + e + h * 8;
        int col = bn + wn * 64 + j * 16 + ln;
        out[(size_t)row * 1024 + col] = acc[i][j][e] + bias[col];
      }
}

// ---------------------------------------------------------------- launcher
extern "C" void kernel_launch(void* const* d_in, const int* in_sizes, int n_in,
                              void* d_out, int out_size, void* d_ws, size_t ws_size,
                              hipStream_t stream) {
  const float* x     = (const float*)d_in[0];  // [4,2048,1024]
  const float* w_qkv = (const float*)d_in[1];  // [1024,3072]
  const float* w_out = (const float*)d_in[2];  // [1024,1024]
  const float* b_out = (const float*)d_in[3];  // [1024]
  float* out = (float*)d_out;                  // [4,2048,1024]

  char* ws = (char*)d_ws;
  bf16* Xb    = (bf16*)(ws);                         // 16 MB  [8192][1024]
  bf16* WqkvT = (bf16*)(ws + 16777216);              //  6 MB  [3072][1024]
  bf16* WoutT = (bf16*)(ws + 23068672);              //  2 MB  [1024][1024]
  bf16* Qd    = (bf16*)(ws + 25165824);              // 16 MB  [64][2048][64]
  bf16* Kd    = (bf16*)(ws + 41943040);              // 16 MB  [64][2048][64]
  bf16* Vt    = (bf16*)(ws + 58720256);              // 16 MB  [64][64][2048]
  bf16* Ob    = (bf16*)(ws + 75497472);              // 16 MB  [8192][1024]

  cvt_kernel<<<8388608 / 256, 256, 0, stream>>>(x, Xb, 8388608);
  tcvt_kernel<<<(1024 * 3072) / 256, 256, 0, stream>>>(w_qkv, WqkvT, 1024, 3072);
  tcvt_kernel<<<(1024 * 1024) / 256, 256, 0, stream>>>(w_out, WoutT, 1024, 1024);

  qkv_gemm_kernel<<<dim3(24, 64), 256, 0, stream>>>(Xb, WqkvT, Qd, Kd, Vt);
  attn_kernel<<<dim3(16, 64), 256, 0, stream>>>(Qd, Kd, Vt, Ob);
  out_gemm_kernel<<<dim3(8, 64), 256, 0, stream>>>(Ob, WoutT, b_out, out);
}
